// QIONFiLMModel_31928786879011
// MI455X (gfx1250) — matrix operand order repaired and checked by reading, not compiled, not run on hardware
//
#include <hip/hip_runtime.h>
#include <hip/hip_bf16.h>
#include <math.h>

// ---------------------------------------------------------------------------
// MI455X (gfx1250) graph-transformer.
// - All large GEMMs on v_wmma_f32_16x16x32_bf16, f32 accumulate.
// - Weights converted once to bf16 AND pre-transposed ([N][K]) so WMMA
//   B-fragments are single contiguous 32B global loads (L2-resident).
// - A tile double-buffered in LDS; global_prefetch_b8 on next B rows.
// - Dense edge bias [B,H,T,T] f32 (256MB ws) built with f32 atomics.
// ---------------------------------------------------------------------------

typedef __bf16 bf16;
typedef __attribute__((ext_vector_type(16))) __bf16 v16bf;
typedef __attribute__((ext_vector_type(8)))  __bf16 v8bf;
typedef __attribute__((ext_vector_type(8)))  float  v8f;

#define B_   32
#define T_   512
#define E_   8192
#define DN_  61
#define DE_  5
#define C_   512
#define H_   8
#define L_   6
#define F_   2048
#define OBS_ 4
#define HID_ 256
#define DH_  64
#define M_   (B_ * T_)   // 16384 rows

__device__ __forceinline__ float gelu_f(float x) {
    return 0.5f * x * (1.0f + erff(x * 0.70710678118654752f));
}

// ------------------- f32 [L][K][N] -> bf16 transposed [L][N][K] ------------
__global__ __launch_bounds__(256) void cvt_bf16_t_kernel(
    const float* __restrict__ s, bf16* __restrict__ d, int K, int N, int L) {
    int idx = blockIdx.x * 256 + threadIdx.x;
    if (idx >= L * K * N) return;
    int l = idx / (K * N);
    int rem = idx - l * (K * N);
    int k = rem / N, n = rem - k * N;          // coalesced read of s
    d[((size_t)l * N + n) * K + k] = (bf16)s[idx];
}

// --------------------------- embed + LayerNorm -----------------------------
__global__ __launch_bounds__(128) void embed_ln_kernel(
    const float* __restrict__ x, const float* __restrict__ W,
    const float* __restrict__ bemb, const float* __restrict__ g,
    const float* __restrict__ bb, float* __restrict__ h) {
    __shared__ float xs[DN_];
    __shared__ float red[128];
    int row = blockIdx.x, tid = threadIdx.x;
    if (tid < DN_) xs[tid] = x[row * DN_ + tid];
    __syncthreads();
    float e[4]; float ps = 0.f;
#pragma unroll
    for (int j = 0; j < 4; ++j) {
        int c = tid * 4 + j;
        float acc = bemb[c];
        for (int k = 0; k < DN_; ++k) acc += xs[k] * W[k * C_ + c];
        e[j] = acc; ps += acc;
    }
    red[tid] = ps; __syncthreads();
    for (int s = 64; s > 0; s >>= 1) { if (tid < s) red[tid] += red[tid + s]; __syncthreads(); }
    float mean = red[0] * (1.0f / C_); __syncthreads();
    float pv = 0.f;
#pragma unroll
    for (int j = 0; j < 4; ++j) { float d0 = e[j] - mean; pv += d0 * d0; }
    red[tid] = pv; __syncthreads();
    for (int s = 64; s > 0; s >>= 1) { if (tid < s) red[tid] += red[tid + s]; __syncthreads(); }
    float rs = rsqrtf(red[0] * (1.0f / C_) + 1e-5f);
#pragma unroll
    for (int j = 0; j < 4; ++j) {
        int c = tid * 4 + j;
        h[(size_t)row * C_ + c] = (e[j] - mean) * rs * g[c] + bb[c];
    }
}

// --------------------------- LayerNorm -> bf16 -----------------------------
__global__ __launch_bounds__(128) void ln_bf16_kernel(
    const float* __restrict__ h, const float* __restrict__ g,
    const float* __restrict__ bb, bf16* __restrict__ y) {
    __shared__ float red[128];
    int row = blockIdx.x, tid = threadIdx.x;
    const float* hr = h + (size_t)row * C_;
    float e[4]; float ps = 0.f;
#pragma unroll
    for (int j = 0; j < 4; ++j) { e[j] = hr[tid * 4 + j]; ps += e[j]; }
    red[tid] = ps; __syncthreads();
    for (int s = 64; s > 0; s >>= 1) { if (tid < s) red[tid] += red[tid + s]; __syncthreads(); }
    float mean = red[0] * (1.0f / C_); __syncthreads();
    float pv = 0.f;
#pragma unroll
    for (int j = 0; j < 4; ++j) { float d0 = e[j] - mean; pv += d0 * d0; }
    red[tid] = pv; __syncthreads();
    for (int s = 64; s > 0; s >>= 1) { if (tid < s) red[tid] += red[tid + s]; __syncthreads(); }
    float rs = rsqrtf(red[0] * (1.0f / C_) + 1e-5f);
#pragma unroll
    for (int j = 0; j < 4; ++j) {
        int c = tid * 4 + j;
        y[(size_t)row * C_ + c] = (bf16)((e[j] - mean) * rs * g[c] + bb[c]);
    }
}

// --------------------------- edge-bias scatter -----------------------------
__global__ __launch_bounds__(256) void edge_bias_kernel(
    const int* __restrict__ ei, const float* __restrict__ ea,
    const float* __restrict__ Web, const float* __restrict__ beb,
    float* __restrict__ bias) {
    int idx = blockIdx.x * 256 + threadIdx.x;
    if (idx >= B_ * E_) return;
    int b = idx / E_, e = idx - b * E_;
    int u = ei[(b * 2 + 0) * E_ + e];
    int v = ei[(b * 2 + 1) * E_ + e];
    float a[DE_];
#pragma unroll
    for (int d = 0; d < DE_; ++d) a[d] = ea[((size_t)b * E_ + e) * DE_ + d];
#pragma unroll
    for (int hh = 0; hh < H_; ++hh) {
        float s = beb[hh];
#pragma unroll
        for (int d = 0; d < DE_; ++d) s += a[d] * Web[d * H_ + hh];
        unsafeAtomicAdd(&bias[(((size_t)b * H_ + hh) * T_ + u) * T_ + v], s);
    }
}

// --------------------------- WMMA GEMM (B pre-transposed) ------------------
// out[M,N] = A[M,K](bf16) @ B[K,N] + bias[N], where Bt = B^T stored [N][K].
// Block 256 thr = 8 waves (2m x 4n), macro tile 128M x 256N, wave tile 64x64.
// A tile double-buffered in LDS; B fragments loaded directly from global/L2.
// EPI 0: -> bf16   EPI 1: gelu -> bf16   EPI 2: residual add -> f32 (outf +=)
template<int EPI>
__global__ __launch_bounds__(256) void gemm_wmma_kernel(
    const bf16* __restrict__ A, const bf16* __restrict__ Bt,
    const float* __restrict__ bias, bf16* __restrict__ outb,
    float* __restrict__ outf, int M, int N, int K) {
    __shared__ bf16 As[2][128][32];
    int tid = threadIdx.x;
    int lane = tid & 31, w = tid >> 5;
    int m0 = blockIdx.x * 128, n0 = blockIdx.y * 256;
    int wm = (w >> 2) * 64;          // 0 / 64
    int wn = (w & 3) * 64;           // 0 / 64 / 128 / 192
    int nl = lane & 15, half = lane >> 4;
    int h8 = half * 8, kh = half * 16;
    int ar = tid >> 2, acg = (tid & 3) * 8;   // A staging: 2 x v8bf / thread

    v8f acc[4][4];
#pragma unroll
    for (int i = 0; i < 4; ++i)
#pragma unroll
        for (int j = 0; j < 4; ++j)
#pragma unroll
            for (int r = 0; r < 8; ++r) acc[i][j][r] = 0.0f;

    // preload k-tile 0
    *(v8bf*)&As[0][ar][acg]      = *(const v8bf*)(A + (size_t)(m0 + ar) * K + acg);
    *(v8bf*)&As[0][ar + 64][acg] = *(const v8bf*)(A + (size_t)(m0 + 64 + ar) * K + acg);
    __syncthreads();

    int nk = K >> 5;
    for (int kt = 0; kt < nk; ++kt) {
        int k0 = kt << 5;
        int cur = kt & 1, nxt = cur ^ 1;
        if (kt + 1 < nk) {   // stage next A tile while computing current
            *(v8bf*)&As[nxt][ar][acg] =
                *(const v8bf*)(A + (size_t)(m0 + ar) * K + k0 + 32 + acg);
            *(v8bf*)&As[nxt][ar + 64][acg] =
                *(const v8bf*)(A + (size_t)(m0 + 64 + ar) * K + k0 + 32 + acg);
        }
        v16bf af[4];
#pragma unroll
        for (int mt = 0; mt < 4; ++mt) {
            int m = wm + mt * 16 + nl;
            v8bf lo = *(const v8bf*)&As[cur][m][kh];
            v8bf hi = *(const v8bf*)&As[cur][m][kh + 8];
#pragma unroll
            for (int i = 0; i < 8; ++i) { af[mt][i] = lo[i]; af[mt][i + 8] = hi[i]; }
        }
#pragma unroll
        for (int nt = 0; nt < 4; ++nt) {
            const bf16* bp = Bt + (size_t)(n0 + wn + nt * 16 + nl) * K + k0 + kh;
            if (kt + 1 < nk) __builtin_prefetch(bp + 32, 0, 3);  // global_prefetch_b8
            v16bf bfr = *(const v16bf*)bp;
#pragma unroll
            for (int mt = 0; mt < 4; ++mt)
                acc[mt][nt] = __builtin_amdgcn_wmma_f32_16x16x32_bf16(
                    false, af[mt], false, bfr, (short)0, acc[mt][nt], false, false);
        }
        __syncthreads();
    }
    // epilogue (C/D layout: col = lane&15, rows 8*half + r)
#pragma unroll
    for (int mt = 0; mt < 4; ++mt)
#pragma unroll
        for (int nt = 0; nt < 4; ++nt) {
            int ncol = n0 + wn + nt * 16 + nl;
            float bv = bias[ncol];
#pragma unroll
            for (int r = 0; r < 8; ++r) {
                int mrow = m0 + wm + mt * 16 + half * 8 + r;
                float vv = acc[mt][nt][r] + bv;
                if (EPI == 1) vv = gelu_f(vv);
                if (EPI == 2) outf[(size_t)mrow * N + ncol] += vv;
                else          outb[(size_t)mrow * N + ncol] = (bf16)vv;
            }
        }
}

// --------------------------- V transpose -----------------------------------
// Vt[b,h,d,t] = qkv[b,t, 2C + h*64 + d]
__global__ __launch_bounds__(256) void vtrans_kernel(const bf16* __restrict__ qkv,
                                                     bf16* __restrict__ Vt) {
    int idx = blockIdx.x * 256 + threadIdx.x;   // (((b*H+h)*DH)+d)*T + t
    int t = idx & (T_ - 1);
    int rest = idx >> 9;
    int d = rest & (DH_ - 1);
    rest >>= 6;
    int hh = rest & (H_ - 1);
    int b = rest >> 3;
    Vt[idx] = qkv[(size_t)(b * T_ + t) * (3 * C_) + 2 * C_ + hh * DH_ + d];
}

// --------------------------- attention (one wave / q-block) ----------------
__global__ __launch_bounds__(32) void attn_kernel(
    const bf16* __restrict__ qkv, const bf16* __restrict__ Vt,
    const float* __restrict__ bias, bf16* __restrict__ o) {
    __shared__ float Sbuf[16][512];
    __shared__ bf16  Pbuf[16][512];
    int blk = blockIdx.x;
    int qb = blk & 31, hh = (blk >> 5) & 7, b = blk >> 8;
    int q0 = qb * 16;
    int lane = threadIdx.x;
    int nl = lane & 15, half = lane >> 4;
    int h8 = half * 8, kh = half * 16;
    const float scale = 0.125f;   // 1/sqrt(Dh)

    // phase 1: S = Q K^T * scale + edge_bias  -> LDS
    const bf16* qbase = qkv + (size_t)(b * T_ + q0 + nl) * (3 * C_) + hh * DH_;
    for (int nb = 0; nb < 32; ++nb) {
        v8f acc;
#pragma unroll
        for (int r = 0; r < 8; ++r) acc[r] = 0.0f;
#pragma unroll
        for (int kk = 0; kk < DH_; kk += 32) {
            v16bf af;
            v8bf lo = *(const v8bf*)(qbase + kk + kh);
            v8bf hi = *(const v8bf*)(qbase + kk + kh + 8);
#pragma unroll
            for (int i = 0; i < 8; ++i) { af[i] = lo[i]; af[i + 8] = hi[i]; }
            const bf16* kbase = qkv + (size_t)(b * T_ + nb * 16 + nl) * (3 * C_)
                                + C_ + hh * DH_ + kk + kh;
            v16bf bfv = *(const v16bf*)kbase;
            acc = __builtin_amdgcn_wmma_f32_16x16x32_bf16(
                false, af, false, bfv, (short)0, acc, false, false);
        }
        const float* bptr = bias + (((size_t)b * H_ + hh) * T_ + q0 + half * 8) * T_
                            + nb * 16 + nl;
#pragma unroll
        for (int r = 0; r < 8; ++r)
            Sbuf[half * 8 + r][nb * 16 + nl] = acc[r] * scale + bptr[(size_t)r * T_];
    }
    __syncthreads();

    // phase 2: softmax over 512 keys (two half-rows per row, shfl-combined)
    {
        int row = nl, c0 = half * 256;
        float mx = -1e30f;
        for (int c = 0; c < 256; ++c) mx = fmaxf(mx, Sbuf[row][c0 + c]);
        mx = fmaxf(mx, __shfl_xor(mx, 16, 32));
        float sum = 0.f;
        for (int c = 0; c < 256; ++c) {
            float ev = __expf(Sbuf[row][c0 + c] - mx);
            Sbuf[row][c0 + c] = ev; sum += ev;
        }
        sum += __shfl_xor(sum, 16, 32);
        float inv = 1.0f / sum;
        for (int c = 0; c < 256; ++c)
            Pbuf[row][c0 + c] = (bf16)(Sbuf[row][c0 + c] * inv);
    }
    __syncthreads();

    // phase 3: O = P @ V  (Vt is [b,h,dh,t] so B fragments are contiguous)
    v8f oacc[4];
#pragma unroll
    for (int nt = 0; nt < 4; ++nt)
#pragma unroll
        for (int r = 0; r < 8; ++r) oacc[nt][r] = 0.0f;
    for (int kc = 0; kc < 16; ++kc) {
        v16bf af;
        v8bf lo = *(const v8bf*)&Pbuf[nl][kc * 32 + kh];
        v8bf hi = *(const v8bf*)&Pbuf[nl][kc * 32 + kh + 8];
#pragma unroll
        for (int i = 0; i < 8; ++i) { af[i] = lo[i]; af[i + 8] = hi[i]; }
#pragma unroll
        for (int nt = 0; nt < 4; ++nt) {
            const bf16* vb = Vt + ((size_t)(b * H_ + hh) * DH_ + nt * 16 + nl) * T_
                             + kc * 32 + kh;
            v16bf bfv = *(const v16bf*)vb;
            oacc[nt] = __builtin_amdgcn_wmma_f32_16x16x32_bf16(
                false, af, false, bfv, (short)0, oacc[nt], false, false);
        }
    }
#pragma unroll
    for (int nt = 0; nt < 4; ++nt) {
        int col = hh * DH_ + nt * 16 + nl;
#pragma unroll
        for (int r = 0; r < 8; ++r)
            o[(size_t)(b * T_ + q0 + half * 8 + r) * C_ + col] = (bf16)oacc[nt][r];
    }
}

// --------------------------- masked mean pool ------------------------------
__global__ __launch_bounds__(256) void pool_kernel(const float* __restrict__ h,
                                                   const unsigned char* __restrict__ mask,
                                                   float* __restrict__ hg) {
    int b = blockIdx.x, tid = threadIdx.x;
    int c0 = tid * 2;
    float s0 = 0.f, s1 = 0.f, cnt = 0.f;
    for (int t = 0; t < T_; ++t) {
        float valid = mask[b * T_ + t] ? 0.f : 1.f;
        cnt += valid;
        const float* row = h + (size_t)(b * T_ + t) * C_;
        s0 += row[c0] * valid; s1 += row[c0 + 1] * valid;
    }
    float inv = 1.0f / (cnt + 1e-9f);
    hg[b * C_ + c0] = s0 * inv;
    hg[b * C_ + c0 + 1] = s1 * inv;
}

// --------------------------- FiLM + MLP head -------------------------------
__global__ __launch_bounds__(256) void head_kernel(
    const float* __restrict__ hg, const float* __restrict__ x_obs,
    const float* __restrict__ g, const float* __restrict__ bb,
    const float* __restrict__ Wf, const float* __restrict__ bf_,
    const float* __restrict__ Wm1, const float* __restrict__ bm1,
    const float* __restrict__ Wm2, const float* __restrict__ bm2,
    const float* __restrict__ Wd1, const float* __restrict__ bd1,
    const float* __restrict__ Wd2, const float* __restrict__ bd2,
    float* __restrict__ out) {
    __shared__ float ht[C_];
    __shared__ float red[256];
    int b = blockIdx.x, tid = threadIdx.x;
    float v0 = hg[b * C_ + tid * 2], v1 = hg[b * C_ + tid * 2 + 1];
    red[tid] = v0 + v1; __syncthreads();
    for (int s = 128; s > 0; s >>= 1) { if (tid < s) red[tid] += red[tid + s]; __syncthreads(); }
    float mean = red[0] * (1.0f / C_); __syncthreads();
    float d0 = v0 - mean, d1 = v1 - mean;
    red[tid] = d0 * d0 + d1 * d1; __syncthreads();
    for (int s = 128; s > 0; s >>= 1) { if (tid < s) red[tid] += red[tid + s]; __syncthreads(); }
    float rs = rsqrtf(red[0] * (1.0f / C_) + 1e-5f);
    float obs[OBS_];
#pragma unroll
    for (int o = 0; o < OBS_; ++o) obs[o] = x_obs[b * OBS_ + o];
#pragma unroll
    for (int j = 0; j < 2; ++j) {
        int c = tid * 2 + j;
        float hn = ((j ? v1 : v0) - mean) * rs * g[c] + bb[c];
        float gam = bf_[c], bet = bf_[C_ + c];
#pragma unroll
        for (int o = 0; o < OBS_; ++o) {
            gam += obs[o] * Wf[o * (2 * C_) + c];
            bet += obs[o] * Wf[o * (2 * C_) + C_ + c];
        }
        ht[c] = (1.0f + gam) * hn + bet;
    }
    __syncthreads();
    float a = bm1[tid];
    for (int k = 0; k < C_; ++k) a += ht[k] * Wm1[k * HID_ + tid];
    float p = gelu_f(a) * Wm2[tid];
    if (tid < HID_ / 2) {
        float d = bd1[tid];
#pragma unroll
        for (int o = 0; o < OBS_; ++o) d += obs[o] * Wd1[o * (HID_ / 2) + tid];
        p += gelu_f(d) * Wd2[tid];
    }
    red[tid] = p; __syncthreads();
    for (int s = 128; s > 0; s >>= 1) { if (tid < s) red[tid] += red[tid + s]; __syncthreads(); }
    if (tid == 0) out[b] = red[0] + bm2[0] + bd2[0];
}

// ---------------------------------------------------------------------------
extern "C" void kernel_launch(void* const* d_in, const int* in_sizes, int n_in,
                              void* d_out, int out_size, void* d_ws, size_t ws_size,
                              hipStream_t stream) {
    (void)in_sizes; (void)n_in; (void)out_size; (void)ws_size;
    const float* x          = (const float*)d_in[0];
    const unsigned char* pm = (const unsigned char*)d_in[1];
    const int*   ei         = (const int*)d_in[2];
    const float* ea         = (const float*)d_in[3];
    const float* xobs       = (const float*)d_in[4];
    const float* W_embed = (const float*)d_in[5];
    const float* b_embed = (const float*)d_in[6];
    const float* pos_g   = (const float*)d_in[7];
    const float* pos_b   = (const float*)d_in[8];
    const float* W_eb    = (const float*)d_in[9];
    const float* b_eb    = (const float*)d_in[10];
    const float* ln1_g   = (const float*)d_in[11];
    const float* ln1_b   = (const float*)d_in[12];
    const float* Wqkv    = (const float*)d_in[13];
    const float* bqkv    = (const float*)d_in[14];
    const float* Wo      = (const float*)d_in[15];
    const float* bo      = (const float*)d_in[16];
    const float* ln2_g   = (const float*)d_in[17];
    const float* ln2_b   = (const float*)d_in[18];
    const float* W1      = (const float*)d_in[19];
    const float* b1      = (const float*)d_in[20];
    const float* W2      = (const float*)d_in[21];
    const float* b2      = (const float*)d_in[22];
    const float* hln_g   = (const float*)d_in[23];
    const float* hln_b   = (const float*)d_in[24];
    const float* W_film  = (const float*)d_in[25];
    const float* b_film  = (const float*)d_in[26];
    const float* Wm1     = (const float*)d_in[27];
    const float* bm1     = (const float*)d_in[28];
    const float* Wm2     = (const float*)d_in[29];
    const float* bm2     = (const float*)d_in[30];
    const float* Wd1     = (const float*)d_in[31];
    const float* bd1     = (const float*)d_in[32];
    const float* Wd2     = (const float*)d_in[33];
    const float* bd2     = (const float*)d_in[34];
    float* out = (float*)d_out;

    size_t off = 0;
    auto carve = [&](size_t bytes) {
        void* p = (char*)d_ws + off;
        off += (bytes + 255) & ~(size_t)255;
        return p;
    };
    float* bias   = (float*)carve((size_t)B_ * H_ * T_ * T_ * 4);   // 256 MB
    float* h      = (float*)carve((size_t)M_ * C_ * 4);             // 32 MB
    bf16*  y      = (bf16*) carve((size_t)M_ * C_ * 2);             // 16 MB
    bf16*  qkv    = (bf16*) carve((size_t)M_ * 3 * C_ * 2);         // 48 MB
    bf16*  Vt     = (bf16*) carve((size_t)B_ * H_ * DH_ * T_ * 2);  // 16 MB
    bf16*  ob     = (bf16*) carve((size_t)M_ * C_ * 2);             // 16 MB
    bf16*  mid    = (bf16*) carve((size_t)M_ * F_ * 2);             // 64 MB
    float* hg     = (float*)carve((size_t)B_ * C_ * 4);
    bf16*  Wqkv_t = (bf16*) carve((size_t)L_ * C_ * 3 * C_ * 2);    // [L][3C][C]
    bf16*  Wo_t   = (bf16*) carve((size_t)L_ * C_ * C_ * 2);        // [L][C][C]
    bf16*  W1_t   = (bf16*) carve((size_t)L_ * C_ * F_ * 2);        // [L][F][C]
    bf16*  W2_t   = (bf16*) carve((size_t)L_ * F_ * C_ * 2);        // [L][C][F]

    // zero edge bias then scatter
    hipMemsetAsync(bias, 0, (size_t)B_ * H_ * T_ * T_ * 4, stream);

    // weight conversion to bf16 with transpose ([L][K][N] -> [L][N][K])
    {
        int n;
        n = L_ * C_ * 3 * C_;
        cvt_bf16_t_kernel<<<(n + 255) / 256, 256, 0, stream>>>(Wqkv, Wqkv_t, C_, 3 * C_, L_);
        n = L_ * C_ * C_;
        cvt_bf16_t_kernel<<<(n + 255) / 256, 256, 0, stream>>>(Wo, Wo_t, C_, C_, L_);
        n = L_ * C_ * F_;
        cvt_bf16_t_kernel<<<(n + 255) / 256, 256, 0, stream>>>(W1, W1_t, C_, F_, L_);
        n = L_ * F_ * C_;
        cvt_bf16_t_kernel<<<(n + 255) / 256, 256, 0, stream>>>(W2, W2_t, F_, C_, L_);
    }

    embed_ln_kernel<<<M_, 128, 0, stream>>>(x, W_embed, b_embed, pos_g, pos_b, h);
    edge_bias_kernel<<<(B_ * E_ + 255) / 256, 256, 0, stream>>>(ei, ea, W_eb, b_eb, bias);

    for (int l = 0; l < L_; ++l) {
        ln_bf16_kernel<<<M_, 128, 0, stream>>>(h, ln1_g + l * C_, ln1_b + l * C_, y);

        gemm_wmma_kernel<0><<<dim3(M_ / 128, (3 * C_) / 256), 256, 0, stream>>>(
            y, Wqkv_t + (size_t)l * C_ * 3 * C_, bqkv + l * 3 * C_,
            qkv, nullptr, M_, 3 * C_, C_);

        vtrans_kernel<<<(B_ * H_ * DH_ * T_) / 256, 256, 0, stream>>>(qkv, Vt);

        attn_kernel<<<B_ * H_ * (T_ / 16), 32, 0, stream>>>(qkv, Vt, bias, ob);

        gemm_wmma_kernel<2><<<dim3(M_ / 128, C_ / 256), 256, 0, stream>>>(
            ob, Wo_t + (size_t)l * C_ * C_, bo + l * C_,
            nullptr, h, M_, C_, C_);

        ln_bf16_kernel<<<M_, 128, 0, stream>>>(h, ln2_g + l * C_, ln2_b + l * C_, y);

        gemm_wmma_kernel<1><<<dim3(M_ / 128, F_ / 256), 256, 0, stream>>>(
            y, W1_t + (size_t)l * C_ * F_, b1 + l * F_,
            mid, nullptr, M_, F_, C_);

        gemm_wmma_kernel<2><<<dim3(M_ / 128, C_ / 256), 256, 0, stream>>>(
            mid, W2_t + (size_t)l * F_ * C_, b2 + l * C_,
            nullptr, h, M_, C_, F_);
    }

    pool_kernel<<<B_, 256, 0, stream>>>(h, pm, hg);
    head_kernel<<<B_, 256, 0, stream>>>(hg, xobs, hln_g, hln_b, W_film, b_film,
                                        Wm1, bm1, Wm2, bm2, Wd1, bd1, Wd2, bd2, out);
}
